// KPConvLayer_2972117369005
// MI455X (gfx1250) — matrix operand order repaired and checked by reading, not compiled, not run on hardware
//
#include <hip/hip_runtime.h>

#define NQ      50000
#define N0      50000
#define MNBR    32
#define KPTS    15
#define DIN     128
#define DOUT    128

typedef __bf16 v16bf __attribute__((ext_vector_type(16)));
typedef __bf16 v2bf  __attribute__((ext_vector_type(2)));
typedef float  v8f   __attribute__((ext_vector_type(8)));

// ---- bf16 conversion: prefer the HW packed converter (v_cvt_pk_bf16_f32) ---
__device__ __forceinline__ unsigned pack_bf16(float lo, float hi) {
#if __has_builtin(__builtin_amdgcn_cvt_pk_bf16_f32)
    v2bf p = __builtin_amdgcn_cvt_pk_bf16_f32(lo, hi);
    return __builtin_bit_cast(unsigned, p);
#else
    v2bf p;
    p.x = (__bf16)lo;
    p.y = (__bf16)hi;
    return __builtin_bit_cast(unsigned, p);
#endif
}
__device__ __forceinline__ unsigned short f2bf(float f) {
#if __has_builtin(__builtin_amdgcn_cvt_pk_bf16_f32)
    v2bf p = __builtin_amdgcn_cvt_pk_bf16_f32(f, f);
    return __builtin_bit_cast(unsigned short, p.x);
#else
    __bf16 h = (__bf16)f;
    return __builtin_bit_cast(unsigned short, h);
#endif
}

// ---------------------------------------------------------------------------
// Prep: convert W [15][128][128] f32 -> bf16 B-fragments for the stage-C GEMM.
// Fragment layout (per out-tile T(8), K-chunk c(60), lane l(32), vgpr v(8)):
//   value = pack( W[kd0][o], W[kd0+1][o] ), kd0 = 32c + 16*(l>>4) + 2v,
//   o = 16T + (l&15);  stored at wf[tid] (tid enumerates exactly this order).
// Also zero-fills a 512B shadow-feature row.
// ---------------------------------------------------------------------------
__global__ void kpconv_prep(const float* __restrict__ W,
                            unsigned* __restrict__ wf,
                            float* __restrict__ zz) {
    int tid = blockIdx.x * 256 + threadIdx.x;          // 480*256 = 122880
    int v  = tid & 7;
    int l  = (tid >> 3) & 31;
    int cT = tid >> 8;                                  // 0..479
    int c  = cT % 60;
    int T  = cT / 60;
    int h  = l >> 4;
    int o  = T * 16 + (l & 15);
    int kd0 = 32 * c + 16 * h + 2 * v;
    int k = kd0 >> 7, d = kd0 & 127;
    const float* p = W + ((size_t)(k * 128 + d)) * 128 + o;
    wf[tid] = pack_bf16(p[0], p[128]);
    if (tid < 128) zz[tid] = 0.0f;
}

// ---------------------------------------------------------------------------
// Main kernel: 1 block = 16 queries, 256 threads = 8 waves (wave32).
// ---------------------------------------------------------------------------
__global__ __launch_bounds__(256)
void kpconv_main(const float* __restrict__ qpts,
                 const float* __restrict__ spts,
                 const int*   __restrict__ nbrs,
                 int is64,
                 const float* __restrict__ x,
                 const float* __restrict__ kpts,
                 const unsigned* __restrict__ wfrag,
                 const float* __restrict__ zerosp,
                 float* __restrict__ out) {
    // LDS
    __shared__ __align__(16) unsigned short w_lds[16 * 16 * 40];   // A-frags (geom weights)
    __shared__ __align__(16) unsigned short wt[16 * 15 * 136];     // stage-B result (bf16)
    __shared__ int   nbr_s[16 * 32];
    __shared__ float qp_s[48];
    __shared__ float kp_s[48];

    const int tid = threadIdx.x;

    // ---- load indices / points -------------------------------------------
    for (int i = tid; i < 512; i += 256) {
        long gi = (long)blockIdx.x * 512 + i;
        nbr_s[i] = is64 ? nbrs[2 * gi] : nbrs[gi];     // little-endian low dword
    }
    if (tid < 48) qp_s[tid] = qpts[(size_t)blockIdx.x * 48 + tid];
    if (tid >= 64 && tid < 64 + 45) kp_s[tid - 64] = kpts[tid - 64];
    __syncthreads();

    // ---- Stage A: geometry weights -> bf16 A-fragments -------------------
    for (int p = tid; p < 512; p += 256) {
        int q = p >> 5, m = p & 31;
        int id = nbr_s[p];
        float px, py, pz;
        if (id >= N0) { px = py = pz = 1.0e6f; }
        else { px = spts[3 * id]; py = spts[3 * id + 1]; pz = spts[3 * id + 2]; }
        float rx = px - qp_s[3 * q];
        float ry = py - qp_s[3 * q + 1];
        float rz = pz - qp_s[3 * q + 2];
        #pragma unroll
        for (int k = 0; k < 15; ++k) {
            float dx = rx - kp_s[3 * k];
            float dy = ry - kp_s[3 * k + 1];
            float dz = rz - kp_s[3 * k + 2];
            float d2 = dx * dx + dy * dy + dz * dz;
            float wv = fmaxf(1.0f - sqrtf(d2) * 10.0f, 0.0f);
            w_lds[q * 640 + k * 40 + m] = f2bf(wv);
        }
        w_lds[q * 640 + 15 * 40 + m] = 0;              // pad row (never consumed)
    }
    __syncthreads();

    const int lane = tid & 31;
    const int wid  = tid >> 5;
    const int cc   = lane & 15;
    const int h    = lane >> 4;

    // ---- Stage B: weighted[q] = Wgeo(16x32) x F(32x128), bf16 WMMA -------
    for (int r = 0; r < 2; ++r) {
        int q = r * 8 + wid;
        union { uint4 u[2]; v16bf v; } A;
        const unsigned short* wrow = &w_lds[q * 640 + cc * 40];
        A.u[0] = *(const uint4*)(wrow + 16 * h);
        A.u[1] = *(const uint4*)(wrow + 16 * h + 8);

        const float* rp[16];
        #pragma unroll
        for (int i = 0; i < 16; ++i) {
            int id = nbr_s[q * 32 + 16 * h + i];
            rp[i] = (id < N0) ? (x + (size_t)id * 128) : zerosp;
        }
        #pragma unroll
        for (int t = 0; t < 8; ++t) {
            union { unsigned u[8]; v16bf v; } B;
            #pragma unroll
            for (int v = 0; v < 8; ++v)
                B.u[v] = pack_bf16(rp[2 * v][16 * t + cc], rp[2 * v + 1][16 * t + cc]);
            v8f zero = {0.f, 0.f, 0.f, 0.f, 0.f, 0.f, 0.f, 0.f};
            v8f D = __builtin_amdgcn_wmma_f32_16x16x32_bf16(
                false, A.v, false, B.v, (short)0, zero, false, false);
            #pragma unroll
            for (int v = 0; v < 8; ++v) {
                int k = v + 8 * h;
                if (k < 15) wt[q * 2040 + k * 136 + 16 * t + cc] = f2bf(D[v]);
            }
        }
    }
    __syncthreads();

    // ---- Stage C: out[16 x 128] = wt[16 x 1920] x Wfrag[1920 x 128] ------
    const int T = wid;
    v8f acc = {0.f, 0.f, 0.f, 0.f, 0.f, 0.f, 0.f, 0.f};
    const unsigned* wfT = wfrag + (size_t)T * 60 * 256;
    #pragma unroll 4
    for (int c = 0; c < 60; ++c) {
        int k  = c >> 2;
        int d0 = (c & 3) * 32 + 16 * h;
        union { uint4 u[2]; v16bf v; } A;
        const unsigned short* wr = &wt[cc * 2040 + k * 136 + d0];  // row = query
        A.u[0] = *(const uint4*)(wr);
        A.u[1] = *(const uint4*)(wr + 8);
        union { uint4 u[2]; v16bf v; } B;
        const unsigned* bp = wfT + c * 256 + lane * 8;
        B.u[0] = *(const uint4*)(bp);
        B.u[1] = *(const uint4*)(bp + 4);
        acc = __builtin_amdgcn_wmma_f32_16x16x32_bf16(
            false, A.v, false, B.v, (short)0, acc, false, false);
    }
    const int o = 16 * T + cc;
    #pragma unroll
    for (int v = 0; v < 8; ++v) {
        int q = v + 8 * h;
        out[((size_t)blockIdx.x * 16 + q) * 128 + o] = acc[v];
    }
}

// ---------------------------------------------------------------------------
extern "C" void kernel_launch(void* const* d_in, const int* in_sizes, int n_in,
                              void* d_out, int out_size, void* d_ws, size_t ws_size,
                              hipStream_t stream) {
    const float* qpts = (const float*)d_in[0];
    const float* spts = (const float*)d_in[1];
    const int*   nbrs = (const int*)d_in[2];
    const float* x    = (const float*)d_in[3];
    const float* kpts = (const float*)d_in[4];
    const float* W    = (const float*)d_in[5];
    float* out        = (float*)d_out;

    int is64 = (in_sizes[2] == 2 * NQ * MNBR) ? 1 : 0;

    unsigned* wf   = (unsigned*)d_ws;                        // 122880 dwords
    float*    zero = (float*)((char*)d_ws + 122880u * 4u);   // 512 B zero row

    kpconv_prep<<<480, 256, 0, stream>>>(W, wf, zero);
    kpconv_main<<<NQ / 16, 256, 0, stream>>>(qpts, spts, nbrs, is64, x, kpts,
                                             wf, zero, out);
}
